// Block_36661840839576
// MI455X (gfx1250) — matrix operand order repaired and checked by reading, not compiled, not run on hardware
//
#include <hip/hip_runtime.h>

// ---------------------------------------------------------------------------
// Transformer block (pre-LN, time-axis LayerNorm) for MI455X / gfx1250.
// Matrix math via v_wmma_f32_16x16x32_f16; A-tile staging via
// global_load_async_to_lds_b128 (ASYNCcnt) on the CDNA5 async path.
// ---------------------------------------------------------------------------

typedef _Float16 half_t;
typedef __attribute__((ext_vector_type(16))) _Float16 v16h;
typedef __attribute__((ext_vector_type(8)))  float    v8f;

#define Bc   2
#define Tc   2048
#define Cc   1024
#define Hc   16
#define HSc  64
#define BTc  (Bc * Tc)
#define ATTN_SCALE 0.03125f   // C**-0.5 = 1/32 (reference scales by full C)
#define LN_EPS 1e-5f

__device__ inline v8f v8f_zero() {
  v8f z;
#pragma unroll
  for (int i = 0; i < 8; ++i) z[i] = 0.0f;
  return z;
}

// Low 32 bits of a generic pointer to __shared__ = LDS byte address.
__device__ inline unsigned lds_u32(const void* p) {
  return (unsigned)(size_t)p;
}

// Async global->LDS 16-byte copy (SADDR form: 64-bit SGPR base + 32-bit
// per-lane unsigned byte offset). Tracked by ASYNCcnt.
__device__ inline void async_copy_b128(unsigned lds_addr, const void* gbase,
                                       unsigned gbyte_off) {
  asm volatile("global_load_async_to_lds_b128 %0, %1, %2"
               :
               : "v"(lds_addr), "v"(gbyte_off), "s"(gbase)
               : "memory");
}

__device__ inline void wait_async0() {
#if __has_builtin(__builtin_amdgcn_s_wait_asynccnt)
  __builtin_amdgcn_s_wait_asynccnt(0);
#else
  asm volatile("s_wait_asynccnt 0" ::: "memory");
#endif
}

// A-fragment (16xK=32, f16). base points at tile origin, rows stride `stride`
// halfs. Lane l: row = l&15, kb=(l>>4)*16, 16 contiguous K values [kb..kb+15]
__device__ inline v16h load_a_frag(const half_t* __restrict__ base, int stride) {
  int lane = threadIdx.x & 31;
  int row = lane & 15;
  int kb  = (lane >> 4) * 16;
  const half_t* p = base + row * stride;
  v16h a;
#pragma unroll
  for (int i = 0; i < 16; ++i) {
    a[i]     = p[kb + i];
  }
  return a;
}

// B-fragment (K=32 x N=16, f16) from "N-major" storage: element (k,n) at
// base[n*nStride + k] (contraction contiguous). Lane l: n = l&15,
// kb = (l>>4)*16, 16 contiguous K values.
__device__ inline v16h load_b_frag(const half_t* __restrict__ base, int nStride) {
  int lane = threadIdx.x & 31;
  int n  = lane & 15;
  int kb = (lane >> 4) * 16;
  const half_t* p = base + n * nStride + kb;
  v16h b;
#pragma unroll
  for (int i = 0; i < 16; ++i) b[i] = p[i];
  return b;
}

__device__ inline v8f wmma_f16(v16h a, v16h b, v8f c) {
  return __builtin_amdgcn_wmma_f32_16x16x32_f16(false, a, false, b, (short)0, c,
                                                false, false);
}

// ---------------------------------------------------------------------------
// Time-axis LayerNorm: per-(b,c) mean/unbiased-var over T, emit f16.
// block = (32 channels, 8 t-slices); grid = (C/32, B)
// ---------------------------------------------------------------------------
__global__ __launch_bounds__(256) void ln_time_kernel(
    const float* __restrict__ x, const float* __restrict__ g,
    const float* __restrict__ beta, half_t* __restrict__ out) {
  __shared__ float s_sum[8][32];
  __shared__ float s_sq[8][32];
  __shared__ float s_mean[32];
  __shared__ float s_rstd[32];
  int tx = threadIdx.x, ty = threadIdx.y;
  int c = blockIdx.x * 32 + tx;
  int b = blockIdx.y;
  const float* xp = x + (size_t)b * Tc * Cc + c;
  float sum = 0.0f, sq = 0.0f;
  for (int t = ty; t < Tc; t += 8) {
    float v = xp[(size_t)t * Cc];
    sum += v; sq += v * v;
  }
  s_sum[ty][tx] = sum; s_sq[ty][tx] = sq;
  __syncthreads();
  if (ty == 0) {
    float S = 0.0f, Q = 0.0f;
#pragma unroll
    for (int i = 0; i < 8; ++i) { S += s_sum[i][tx]; Q += s_sq[i][tx]; }
    float m = S / (float)Tc;
    float var = (Q - (float)Tc * m * m) / (float)(Tc - 1);  // ddof=1
    s_mean[tx] = m;
    s_rstd[tx] = rsqrtf(var + LN_EPS);
  }
  __syncthreads();
  float m = s_mean[tx], r = s_rstd[tx], gg = g[c], bb = beta[c];
  half_t* op = out + (size_t)b * Tc * Cc + c;
  for (int t = ty; t < Tc; t += 8) {
    float v = xp[(size_t)t * Cc];
    op[(size_t)t * Cc] = (half_t)(gg * (v - m) * r + bb);
  }
}

// ---------------------------------------------------------------------------
// QKV projection: per (b,h), (T x C) f16 @ (C x 64) f32-weights -> q,k (f16,
// T x 64 row-major) and v transposed (64 x T) for contiguous PV B-fragments.
// grid = (T/128, 1, B*H); block = 256 (8 waves, 4x2, wave tile 32x32)
// ---------------------------------------------------------------------------
__global__ __launch_bounds__(256) void qkv_kernel(
    const half_t* __restrict__ h1, const float* __restrict__ Wq,
    const float* __restrict__ Wk, const float* __restrict__ Wv,
    half_t* __restrict__ qo, half_t* __restrict__ ko,
    half_t* __restrict__ vTo) {
  __shared__ half_t sA[128 * 32];
  __shared__ half_t sBq[64 * 32];   // transposed [n][k]
  __shared__ half_t sBk[64 * 32];
  __shared__ half_t sBv[64 * 32];
  int tid = threadIdx.x;
  int z = blockIdx.z;
  int b = z >> 4, h = z & 15;
  int m0 = blockIdx.x * 128;
  const half_t* A  = h1 + (size_t)b * Tc * Cc;
  const float*  wq = Wq + (size_t)h * Cc * HSc;
  const float*  wk = Wk + (size_t)h * Cc * HSc;
  const float*  wv = Wv + (size_t)h * Cc * HSc;
  int w = tid >> 5, lane = tid & 31;
  int wm = w >> 1, wn = w & 1;

  v8f aq[2][2], ak[2][2], av[2][2];
#pragma unroll
  for (int i = 0; i < 2; ++i)
#pragma unroll
    for (int j = 0; j < 2; ++j) { aq[i][j] = v8f_zero(); ak[i][j] = v8f_zero(); av[i][j] = v8f_zero(); }

  int arow = tid >> 1, akoff = (tid & 1) * 16;
  unsigned aLds = lds_u32(sA + arow * 32 + akoff);

  for (int kk = 0; kk < Cc; kk += 32) {
    // A tile 128x32 f16: async global->LDS (1 x 16B per thread)
    {
      unsigned goff = (unsigned)(((size_t)(m0 + arow) * Cc + kk + akoff) *
                                 sizeof(half_t));
      async_copy_b128(aLds, A, goff);
    }
    {  // B tiles 32x64 f32 -> f16, stored transposed [n][k]
      int k = tid >> 3, nb = (tid & 7) * 8;
      size_t gi = (size_t)(kk + k) * HSc + nb;
#pragma unroll
      for (int j = 0; j < 8; ++j) {
        sBq[(nb + j) * 32 + k] = (half_t)wq[gi + j];
        sBk[(nb + j) * 32 + k] = (half_t)wk[gi + j];
        sBv[(nb + j) * 32 + k] = (half_t)wv[gi + j];
      }
    }
    wait_async0();
    __syncthreads();
    v16h af[2];
#pragma unroll
    for (int i = 0; i < 2; ++i) af[i] = load_a_frag(sA + (wm * 32 + i * 16) * 32, 32);
#pragma unroll
    for (int j = 0; j < 2; ++j) {
      int nb = (wn * 32 + j * 16) * 32;
      v16h bq = load_b_frag(sBq + nb, 32);
      v16h bk = load_b_frag(sBk + nb, 32);
      v16h bv = load_b_frag(sBv + nb, 32);
#pragma unroll
      for (int i = 0; i < 2; ++i) {
        aq[i][j] = wmma_f16(af[i], bq, aq[i][j]);
        ak[i][j] = wmma_f16(af[i], bk, ak[i][j]);
        av[i][j] = wmma_f16(af[i], bv, av[i][j]);
      }
    }
    __syncthreads();
  }
  int hi = lane >> 4, col = lane & 15;
#pragma unroll
  for (int i = 0; i < 2; ++i)
#pragma unroll
    for (int j = 0; j < 2; ++j)
#pragma unroll
      for (int r = 0; r < 8; ++r) {
        int row = m0 + wm * 32 + i * 16 + r + 8 * hi;
        int n = wn * 32 + j * 16 + col;
        qo[((size_t)z * Tc + row) * HSc + n]  = (half_t)aq[i][j][r];
        ko[((size_t)z * Tc + row) * HSc + n]  = (half_t)ak[i][j][r];
        vTo[((size_t)z * HSc + n) * Tc + row] = (half_t)av[i][j][r];
      }
}

// ---------------------------------------------------------------------------
// Causal flash attention. One wave per 16 q-rows; key blocks of 32.
// grid = (T/128, H, B); block = 256 (8 waves)
// ---------------------------------------------------------------------------
__global__ __launch_bounds__(256) void attn_kernel(
    const half_t* __restrict__ q, const half_t* __restrict__ k,
    const half_t* __restrict__ vT, half_t* __restrict__ attn) {
  __shared__ half_t sP[8][16 * 32];  // per-wave P scratch (C-layout -> A-layout)
  int tid = threadIdx.x, w = tid >> 5, lane = tid & 31;
  int b = blockIdx.z, h = blockIdx.y;
  int z = b * Hc + h;
  int qb = blockIdx.x * 8 + w;
  const half_t* qp = q  + (size_t)z * Tc * HSc;
  const half_t* kp = k  + (size_t)z * Tc * HSc;
  const half_t* vp = vT + (size_t)z * HSc * Tc;
  int hi = lane >> 4, col = lane & 15;

  // q fragments: two 16x32 A tiles covering d=0..31 and d=32..63
  v16h qf[2];
#pragma unroll
  for (int c = 0; c < 2; ++c) qf[c] = load_a_frag(qp + (size_t)(qb * 16) * HSc + c * 32, HSc);

  v8f O[4];
#pragma unroll
  for (int j = 0; j < 4; ++j) O[j] = v8f_zero();
  float mrow[8], lrow[8];
#pragma unroll
  for (int r = 0; r < 8; ++r) { mrow[r] = -__builtin_inff(); lrow[r] = 0.0f; }

  int qmax = qb * 16 + 15;
  half_t* sp = sP[w];

  for (int kb0 = 0; kb0 <= qmax; kb0 += 32) {
    // S = q . k^T (two 16x16 tiles), K=64 split into 2 WMMAs each
    v8f s[2];
#pragma unroll
    for (int j = 0; j < 2; ++j) {
      s[j] = v8f_zero();
#pragma unroll
      for (int c = 0; c < 2; ++c) {
        v16h kf = load_b_frag(kp + (size_t)(kb0 + j * 16) * HSc + c * 32, HSc);
        s[j] = wmma_f16(qf[c], kf, s[j]);
      }
    }
    // scale + causal mask + online softmax
    float p0[8], p1[8], rmax[8];
#pragma unroll
    for (int r = 0; r < 8; ++r) {
      int qrow = qb * 16 + r + 8 * hi;
      float v0 = s[0][r] * ATTN_SCALE;
      float v1 = s[1][r] * ATTN_SCALE;
      if (kb0 + col > qrow)      v0 = -__builtin_inff();
      if (kb0 + 16 + col > qrow) v1 = -__builtin_inff();
      p0[r] = v0; p1[r] = v1;
      rmax[r] = fmaxf(v0, v1);
    }
#pragma unroll
    for (int off = 1; off < 16; off <<= 1)
#pragma unroll
      for (int r = 0; r < 8; ++r) rmax[r] = fmaxf(rmax[r], __shfl_xor(rmax[r], off, 32));
    float rsum[8];
#pragma unroll
    for (int r = 0; r < 8; ++r) {
      float mn = fmaxf(mrow[r], rmax[r]);
      float alpha = __expf(mrow[r] - mn);
      mrow[r] = mn;
      p0[r] = __expf(p0[r] - mn);
      p1[r] = __expf(p1[r] - mn);
      rsum[r] = p0[r] + p1[r];
      lrow[r] = lrow[r] * alpha;
#pragma unroll
      for (int j = 0; j < 4; ++j) O[j][r] *= alpha;
    }
#pragma unroll
    for (int off = 1; off < 16; off <<= 1)
#pragma unroll
      for (int r = 0; r < 8; ++r) rsum[r] += __shfl_xor(rsum[r], off, 32);
#pragma unroll
    for (int r = 0; r < 8; ++r) lrow[r] += rsum[r];

    // P (C-layout) -> LDS -> A-fragment
#pragma unroll
    for (int r = 0; r < 8; ++r) {
      int rr = r + 8 * hi;
      sp[rr * 32 + col]      = (half_t)p0[r];
      sp[rr * 32 + 16 + col] = (half_t)p1[r];
    }
    v16h pf = load_a_frag(sp, 32);
    // O += P . V  (V via transposed layout: contraction contiguous)
#pragma unroll
    for (int j = 0; j < 4; ++j) {
      v16h vf = load_b_frag(vp + (size_t)(j * 16) * Tc + kb0, Tc);
      O[j] = wmma_f16(pf, vf, O[j]);
    }
  }

  // normalize and store concat-head layout: attn[(b*T + t)*C + h*64 + d]
#pragma unroll
  for (int r = 0; r < 8; ++r) {
    float inv = 1.0f / lrow[r];
    int t = qb * 16 + r + 8 * hi;
#pragma unroll
    for (int j = 0; j < 4; ++j) {
      attn[((size_t)b * Tc + t) * Cc + h * HSc + j * 16 + col] = (half_t)(O[j][r] * inv);
    }
  }
}

// ---------------------------------------------------------------------------
// Generic WMMA GEMM: C = A(f16, MxK) @ B(f32->f16, KxN) with epilogue
// (bias, residual f32, relu, f32 and/or f16 outputs).
// block tile 128x128, BK=32; 8 waves 4x2, wave tile 32x64.
// A tiles staged with async global->LDS copies.
// ---------------------------------------------------------------------------
__global__ __launch_bounds__(256) void gemm_kernel(
    const half_t* __restrict__ A, const float* __restrict__ Bm,
    const float* __restrict__ bias, const float* __restrict__ resid,
    float* __restrict__ outf, half_t* __restrict__ outh,
    int M, int N, int K, int relu) {
  __shared__ half_t sA[128 * 32];
  __shared__ half_t sBt[128 * 32];  // transposed [n][k]
  int tid = threadIdx.x, w = tid >> 5, lane = tid & 31;
  int wm = w >> 1, wn = w & 1;
  int m0 = blockIdx.x * 128, n0 = blockIdx.y * 128;

  v8f acc[2][4];
#pragma unroll
  for (int i = 0; i < 2; ++i)
#pragma unroll
    for (int j = 0; j < 4; ++j) acc[i][j] = v8f_zero();

  int arow = tid >> 1, akoff = (tid & 1) * 16;
  unsigned aLds = lds_u32(sA + arow * 32 + akoff);

  for (int kk = 0; kk < K; kk += 32) {
    // A tile: async global->LDS 16B per thread
    {
      unsigned goff = (unsigned)(((size_t)(m0 + arow) * K + kk + akoff) *
                                 sizeof(half_t));
      async_copy_b128(aLds, A, goff);
    }
    {  // B tile 32x128 f32 -> f16 transposed
      int k = tid >> 3, nb = (tid & 7) * 16;
      size_t gi = (size_t)(kk + k) * N + n0 + nb;
#pragma unroll
      for (int j = 0; j < 16; ++j) sBt[(nb + j) * 32 + k] = (half_t)Bm[gi + j];
    }
    wait_async0();
    __syncthreads();
    v16h af[2];
#pragma unroll
    for (int i = 0; i < 2; ++i) af[i] = load_a_frag(sA + (wm * 32 + i * 16) * 32, 32);
#pragma unroll
    for (int j = 0; j < 4; ++j) {
      v16h bf = load_b_frag(sBt + (wn * 64 + j * 16) * 32, 32);
#pragma unroll
      for (int i = 0; i < 2; ++i) acc[i][j] = wmma_f16(af[i], bf, acc[i][j]);
    }
    __syncthreads();
  }

  int hi = lane >> 4, col = lane & 15;
#pragma unroll
  for (int i = 0; i < 2; ++i)
#pragma unroll
    for (int j = 0; j < 4; ++j)
#pragma unroll
      for (int r = 0; r < 8; ++r) {
        int row = m0 + wm * 32 + i * 16 + r + 8 * hi;
        int c = n0 + wn * 64 + j * 16 + col;
        float v = acc[i][j][r];
        if (bias)  v += bias[c];
        if (resid) v += resid[(size_t)row * N + c];
        if (relu)  v = fmaxf(v, 0.0f);
        if (outf) outf[(size_t)row * N + c] = v;
        if (outh) outh[(size_t)row * N + c] = (half_t)v;
      }
}

// ---------------------------------------------------------------------------
// Host launcher
// ---------------------------------------------------------------------------
extern "C" void kernel_launch(void* const* d_in, const int* in_sizes, int n_in,
                              void* d_out, int out_size, void* d_ws, size_t ws_size,
                              hipStream_t stream) {
  (void)in_sizes; (void)n_in; (void)out_size; (void)ws_size;
  const float* x     = (const float*)d_in[0];
  const float* Wq    = (const float*)d_in[1];
  const float* Wk    = (const float*)d_in[2];
  const float* Wv    = (const float*)d_in[3];
  const float* Wproj = (const float*)d_in[4];
  const float* bproj = (const float*)d_in[5];
  const float* W1    = (const float*)d_in[6];
  const float* b1    = (const float*)d_in[7];
  const float* W2    = (const float*)d_in[8];
  const float* b2    = (const float*)d_in[9];
  const float* g1    = (const float*)d_in[10];
  const float* beta1 = (const float*)d_in[11];
  const float* g2    = (const float*)d_in[12];
  const float* beta2 = (const float*)d_in[13];

  const size_t MB = 1024ull * 1024ull;
  char* ws = (char*)d_ws;
  half_t* h1   = (half_t*)(ws + 0 * MB);    // B*T*C f16  (8 MB)
  half_t* qb_  = (half_t*)(ws + 8 * MB);    // B*H*T*64 f16
  half_t* kb_  = (half_t*)(ws + 16 * MB);
  half_t* vT   = (half_t*)(ws + 24 * MB);   // B*H*64*T f16 (transposed)
  half_t* attn = (half_t*)(ws + 32 * MB);   // B*T*C f16
  half_t* h2   = (half_t*)(ws + 40 * MB);
  half_t* y    = (half_t*)(ws + 48 * MB);
  float*  x1   = (float*)(ws + 56 * MB);    // B*T*C f32 (16 MB)

  dim3 lnBlk(32, 8);
  dim3 lnGrid(Cc / 32, Bc);

  // 1) h1 = ln1(x) in f16
  ln_time_kernel<<<lnGrid, lnBlk, 0, stream>>>(x, g1, beta1, h1);

  // 2) q, k, vT
  qkv_kernel<<<dim3(Tc / 128, 1, Bc * Hc), 256, 0, stream>>>(h1, Wq, Wk, Wv, qb_, kb_, vT);

  // 3) causal flash attention -> attn (f16, concat heads)
  attn_kernel<<<dim3(Tc / 128, Hc, Bc), 256, 0, stream>>>(qb_, kb_, vT, attn);

  // 4) x1 = x + attn @ Wproj + bproj
  gemm_kernel<<<dim3(BTc / 128, Cc / 128), 256, 0, stream>>>(
      attn, Wproj, bproj, x, x1, nullptr, BTc, Cc, Cc, 0);

  // 5) h2 = ln2(x1) in f16
  ln_time_kernel<<<lnGrid, lnBlk, 0, stream>>>(x1, g2, beta2, h2);

  // 6) y = relu(h2 @ W1 + b1) in f16
  gemm_kernel<<<dim3(BTc / 128, Cc / 128), 256, 0, stream>>>(
      h2, W1, b1, nullptr, nullptr, y, BTc, Cc, Cc, 1);

  // 7) out = x1 + y @ W2 + b2 (f32)
  gemm_kernel<<<dim3(BTc / 128, Cc / 128), 256, 0, stream>>>(
      y, W2, b2, x1, (float*)d_out, nullptr, BTc, Cc, Cc, 0);
}
